// Net_34248069219045
// MI455X (gfx1250) — matrix operand order repaired and checked by reading, not compiled, not run on hardware
//
#include <hip/hip_runtime.h>
#include <hip/hip_fp16.h>
#include <math.h>

// ---------------- problem constants (match reference) ----------------
#define NN    65536      // nodes
#define EE    524288     // edges
#define FIN   128        // input features
#define NH2   32         // nhid/2
#define HCC   192        // heads*channels
#define NHEAD 3
#define CPH   64         // channels per head
#define NB    128        // graphs
#define NPG   512        // nodes per graph
#define KKEEP 256        // ceil(0.5*512)
#define NCLS  10

typedef __attribute__((ext_vector_type(16))) _Float16 v16h;
typedef __attribute__((ext_vector_type(8)))  float    v8f;

// ---- WMMA fragment index maps (per CDNA5 ISA 7.12.2, wave32) ----
// A (16-bit, 16x32 MxK): lane r=lane&15 -> M=r ; g=lane>>4 ; element e (0..15):
//   K = (e<8 ? e : e+8) + 8*g
__device__ __forceinline__ int amapK(int e, int g) { return (e < 8 ? e : e + 8) + 8 * g; }
// B (16-bit, 32x16 KxN): N = lane&15 ; element e: K = (e<8 ? e : e+8) + 8*g
__device__ __forceinline__ int bmapK(int e, int g) { return (e < 8 ? e : e + 8) + 8 * g; }
// C/D f32 16x16: VGPR v: M = v + 8*g ; N = lane&15

__device__ __forceinline__ void atomicMaxF(float* addr, float val) {
    unsigned* ua = (unsigned*)addr;
    unsigned old = __float_as_uint(*addr);
    while (__uint_as_float(old) < val) {
        unsigned assumed = old;
        old = atomicCAS(ua, assumed, __float_as_uint(val));
        if (old == assumed) break;
    }
}

// ---------------- generic fill ----------------
__global__ void k_fill(float* __restrict__ p, float v, int n) {
    int i = blockIdx.x * blockDim.x + threadIdx.x;
    if (i < n) p[i] = v;
}

// ---------------- 1) h = ELU(x @ W_lin^T + b_lin), stored f16 [N,32] ----------------
// one wave per 16-node tile; 2 accumulators (cols 0-15 / 16-31); K=128 in 4 steps
__global__ void k_lin_elu(const float* __restrict__ x, const float* __restrict__ Wlin,
                          const float* __restrict__ blin, _Float16* __restrict__ h) {
    int lane = threadIdx.x & 31;
    int wv   = blockIdx.x * (blockDim.x >> 5) + (threadIdx.x >> 5);
    int row0 = wv * 16;
    int g = lane >> 4, r = lane & 15;
    v8f c0 = {}, c1 = {};
    for (int kk = 0; kk < FIN; kk += 32) {
        v16h a, b0, b1;
#pragma unroll
        for (int e = 0; e < 16; ++e) {
            a[e]  = (_Float16)x[(row0 + r) * FIN + kk + amapK(e, g)];
            int kb = kk + bmapK(e, g);
            b0[e] = (_Float16)Wlin[r * FIN + kb];          // B[k][n] = Wlin[n][k]
            b1[e] = (_Float16)Wlin[(r + 16) * FIN + kb];
        }
        c0 = __builtin_amdgcn_wmma_f32_16x16x32_f16(false, a, false, b0, (short)0, c0, false, false);
        c1 = __builtin_amdgcn_wmma_f32_16x16x32_f16(false, a, false, b1, (short)0, c1, false, false);
    }
#pragma unroll
    for (int v = 0; v < 8; ++v) {
        int m = v + 8 * g;
        float y0 = c0[v] + blin[r];
        float y1 = c1[v] + blin[r + 16];
        y0 = y0 > 0.f ? y0 : (__expf(y0) - 1.f);
        y1 = y1 > 0.f ? y1 : (__expf(y1) - 1.f);
        _Float16* hp = h + (size_t)(row0 + m) * NH2;
        hp[r]      = (_Float16)y0;
        hp[r + 16] = (_Float16)y1;
    }
}

// ---------------- 2) xp = h @ W_src^T  [N,192] f32 ----------------
// wave per (16-node tile, 32-col group); K=32 -> single WMMA per accumulator
__global__ void k_xp(const _Float16* __restrict__ h, const float* __restrict__ Wsrc,
                     float* __restrict__ xp) {
    int lane = threadIdx.x & 31;
    int wv   = blockIdx.x * (blockDim.x >> 5) + (threadIdx.x >> 5);
    int tile = wv / 6, cg = wv % 6;
    int row0 = tile * 16, col0 = cg * 32;
    int g = lane >> 4, r = lane & 15;
    v16h a, b0, b1;
#pragma unroll
    for (int e = 0; e < 16; ++e) {
        a[e]  = h[(size_t)(row0 + r) * NH2 + amapK(e, g)];
        int kb = bmapK(e, g);
        b0[e] = (_Float16)Wsrc[(col0 + r) * NH2 + kb];
        b1[e] = (_Float16)Wsrc[(col0 + r + 16) * NH2 + kb];
    }
    v8f c0 = {}, c1 = {};
    c0 = __builtin_amdgcn_wmma_f32_16x16x32_f16(false, a, false, b0, (short)0, c0, false, false);
    c1 = __builtin_amdgcn_wmma_f32_16x16x32_f16(false, a, false, b1, (short)0, c1, false, false);
#pragma unroll
    for (int v = 0; v < 8; ++v) {
        int m = v + 8 * g;
        float* op = xp + (size_t)(row0 + m) * HCC + col0;
        op[r]      = c0[v];
        op[r + 16] = c1[v];
    }
}

// ---------------- 3) per-node attention scalars a_src/a_dst [N,3] ----------------
__global__ void k_attn_node(const float* __restrict__ xp, const float* __restrict__ att_src,
                            const float* __restrict__ att_dst,
                            float* __restrict__ asv, float* __restrict__ adv) {
    int i = blockIdx.x * blockDim.x + threadIdx.x;
    if (i >= NN * NHEAD) return;
    int n = i / NHEAD, hd = i % NHEAD;
    const float* xr = xp + (size_t)n * HCC + hd * CPH;
    const float* as = att_src + hd * CPH;
    const float* ad = att_dst + hd * CPH;
    float s0 = 0.f, s1 = 0.f;
    for (int c = 0; c < CPH; ++c) { float v = xr[c]; s0 += v * as[c]; s1 += v * ad[c]; }
    asv[i] = s0; adv[i] = s1;
}

// ---------------- 4) in-degree count + self-loop edge_attr mean ----------------
__global__ void k_cnt_loop(const int* __restrict__ dst, const float* __restrict__ ea,
                           float* __restrict__ cnt, float* __restrict__ la) {
    int e = blockIdx.x * blockDim.x + threadIdx.x;
    if (e >= EE) return;
    int d = dst[e];
    atomicAdd(&cnt[d], 1.f);
    atomicAdd(&la[d * 2 + 0], ea[e * 2 + 0]);
    atomicAdd(&la[d * 2 + 1], ea[e * 2 + 1]);
}
__global__ void k_loop_norm(const float* __restrict__ cnt, float* __restrict__ la) {
    int n = blockIdx.x * blockDim.x + threadIdx.x;
    if (n >= NN) return;
    float inv = 1.f / fmaxf(cnt[n], 1.f);
    la[n * 2 + 0] *= inv;
    la[n * 2 + 1] *= inv;
}

// edge-attr attention is linear: a_edge(hd) = ea0*u0[hd] + ea1*u1[hd]
__global__ void k_uedge(const float* __restrict__ Wedge, const float* __restrict__ att_edge,
                        float* __restrict__ uf) {
    int t = threadIdx.x;
    if (t >= NHEAD * 2) return;
    int hd = t >> 1, j = t & 1;
    float s = 0.f;
    for (int c = 0; c < CPH; ++c) s += Wedge[(hd * CPH + c) * 2 + j] * att_edge[hd * CPH + c];
    uf[hd * 2 + j] = s;
}

// ---------------- 5) alpha = leaky_relu(asrc[s]+adst[d]+aedge), segment-max ----------------
__global__ void k_alpha(const int* __restrict__ src, const int* __restrict__ dst,
                        const float* __restrict__ ea, const float* __restrict__ la,
                        const float* __restrict__ asv, const float* __restrict__ adv,
                        const float* __restrict__ uf,
                        float* __restrict__ alpha, float* __restrict__ mseg) {
    int idx = blockIdx.x * blockDim.x + threadIdx.x;
    if (idx >= EE + NN) return;
    int s, d; float e0, e1;
    if (idx < EE) { s = src[idx]; d = dst[idx]; e0 = ea[2 * idx]; e1 = ea[2 * idx + 1]; }
    else { int n = idx - EE; s = n; d = n; e0 = la[2 * n]; e1 = la[2 * n + 1]; }
#pragma unroll
    for (int hd = 0; hd < NHEAD; ++hd) {
        float al = asv[s * NHEAD + hd] + adv[d * NHEAD + hd] + e0 * uf[hd * 2] + e1 * uf[hd * 2 + 1];
        al = al > 0.f ? al : 0.2f * al;
        alpha[(size_t)idx * NHEAD + hd] = al;
        atomicMaxF(&mseg[d * NHEAD + hd], al);
    }
}

// ---------------- 6) ealpha = exp(alpha - m[d]); denom += ----------------
__global__ void k_expsum(const int* __restrict__ dst, const float* __restrict__ mseg,
                         float* __restrict__ alpha, float* __restrict__ denom) {
    int i = blockIdx.x * blockDim.x + threadIdx.x;
    if (i >= (EE + NN) * NHEAD) return;
    int idx = i / NHEAD, hd = i % NHEAD;
    int d = (idx < EE) ? dst[idx] : (idx - EE);
    float e = __expf(alpha[i] - mseg[d * NHEAD + hd]);
    alpha[i] = e;
    atomicAdd(&denom[d * NHEAD + hd], e);
}

// ---------------- 7) out[d] += (ealpha/denom) * xp[s]  (4 channels / thread) ----------------
__global__ void k_agg(const int* __restrict__ src, const int* __restrict__ dst,
                      const float* __restrict__ alpha, const float* __restrict__ denom,
                      const float* __restrict__ xp, float* __restrict__ outacc) {
    long long i = (long long)blockIdx.x * blockDim.x + threadIdx.x;
    if (i >= (long long)(EE + NN) * 48) return;
    int idx = (int)(i / 48), q = (int)(i % 48);
    int c0 = q * 4, hd = c0 / CPH;
    int s, d;
    if (idx < EE) { s = src[idx]; d = dst[idx]; } else { s = idx - EE; d = s; }
    float w = alpha[(size_t)idx * NHEAD + hd] / (denom[d * NHEAD + hd] + 1e-16f);
    const float* xs = xp + (size_t)s * HCC + c0;
    float* od = outacc + (size_t)d * HCC + c0;
#pragma unroll
    for (int j = 0; j < 4; ++j) atomicAdd(&od[j], w * xs[j]);
}

// ---------------- 8) x1 = relu(out + b_gat)  (in place) ----------------
__global__ void k_x1(float* __restrict__ outacc, const float* __restrict__ bgat) {
    int i = blockIdx.x * blockDim.x + threadIdx.x;
    if (i >= NN * HCC) return;
    outacc[i] = fmaxf(outacc[i] + bgat[i % HCC], 0.f);
}

// ---------------- 9) GCN score: xw, dinv; scatter; final ----------------
__global__ void k_score_prep(const float* __restrict__ x1, const float* __restrict__ Wgcn,
                             const float* __restrict__ cnt,
                             float* __restrict__ xw, float* __restrict__ dinv) {
    int n = blockIdx.x * blockDim.x + threadIdx.x;
    if (n >= NN) return;
    const float* xr = x1 + (size_t)n * HCC;
    float s = 0.f;
    for (int c = 0; c < HCC; ++c) s += xr[c] * Wgcn[c];
    xw[n] = s;
    dinv[n] = rsqrtf(cnt[n] + 1.f);
}
__global__ void k_gcn_scatter(const int* __restrict__ src, const int* __restrict__ dst,
                              const float* __restrict__ xw, const float* __restrict__ dinv,
                              float* __restrict__ agg) {
    int e = blockIdx.x * blockDim.x + threadIdx.x;
    if (e >= EE) return;
    int s = src[e], d = dst[e];
    atomicAdd(&agg[d], dinv[s] * dinv[d] * xw[s]);
}
__global__ void k_score_final(const float* __restrict__ agg, const float* __restrict__ xw,
                              const float* __restrict__ dinv, const float* __restrict__ bgcn,
                              float* __restrict__ score) {
    int n = blockIdx.x * blockDim.x + threadIdx.x;
    if (n >= NN) return;
    score[n] = agg[n] + dinv[n] * dinv[n] * xw[n] + bgcn[0];
}

// ---------------- 10) per-graph top-k mask + gmp/gap readout ----------------
__global__ void k_pool(const float* __restrict__ score, const float* __restrict__ x1,
                       float* __restrict__ r) {
    __shared__ float sc[NPG];
    __shared__ float th[NPG];
    __shared__ unsigned char msk[NPG];
    int g = blockIdx.x, tid = threadIdx.x;
    for (int j = tid; j < NPG; j += blockDim.x) {
        float s = score[g * NPG + j];
        sc[j] = s;
        th[j] = tanhf(s);
    }
    __syncthreads();
    // stable descending rank: count (s_t > s_j) || (s_t == s_j && t < j)
    for (int j = tid; j < NPG; j += blockDim.x) {
        float sj = sc[j];
        int c = 0;
        for (int t = 0; t < NPG; ++t) {
            float st = sc[t];
            c += (st > sj) || (st == sj && t < j);
        }
        msk[j] = (c < KKEEP) ? 1 : 0;
    }
    __syncthreads();
    for (int c = tid; c < HCC; c += blockDim.x) {
        float mx = -3.0e38f, sm = 0.f;
        for (int j = 0; j < NPG; ++j) {
            if (msk[j]) {
                float v = x1[((size_t)g * NPG + j) * HCC + c] * th[j];
                mx = fmaxf(mx, v);
                sm += v;
            }
        }
        r[g * (2 * HCC) + c]       = mx;
        r[g * (2 * HCC) + HCC + c] = sm * (1.f / KKEEP);
    }
}

// ---------------- 11) classifier MLP + log_softmax (block per graph) ----------------
__global__ void k_mlp(const float* __restrict__ r,
                      const float* __restrict__ W1, const float* __restrict__ b1,
                      const float* __restrict__ W2, const float* __restrict__ b2,
                      const float* __restrict__ W3, const float* __restrict__ b3,
                      float* __restrict__ out) {
    __shared__ float rin[2 * HCC];
    __shared__ float o1[64];
    __shared__ float o2[32];
    __shared__ float o3[NCLS];
    int g = blockIdx.x, t = threadIdx.x;
    for (int j = t; j < 2 * HCC; j += blockDim.x) rin[j] = r[g * (2 * HCC) + j];
    __syncthreads();
    if (t < 64) {
        float a = b1[t];
        for (int j = 0; j < 2 * HCC; ++j) a += W1[t * (2 * HCC) + j] * rin[j];
        o1[t] = fmaxf(a, 0.f);
    }
    __syncthreads();
    if (t < 32) {
        float a = b2[t];
        for (int j = 0; j < 64; ++j) a += W2[t * 64 + j] * o1[j];
        o2[t] = fmaxf(a, 0.f);
    }
    __syncthreads();
    if (t < NCLS) {
        float a = b3[t];
        for (int j = 0; j < 32; ++j) a += W3[t * 32 + j] * o2[j];
        o3[t] = a;
    }
    __syncthreads();
    if (t == 0) {
        float mx = o3[0];
        for (int j = 1; j < NCLS; ++j) mx = fmaxf(mx, o3[j]);
        float sm = 0.f;
        for (int j = 0; j < NCLS; ++j) sm += __expf(o3[j] - mx);
        float lse = mx + logf(sm);
        for (int j = 0; j < NCLS; ++j) out[g * NCLS + j] = o3[j] - lse;
    }
}

// ===================== host launcher =====================
extern "C" void kernel_launch(void* const* d_in, const int* in_sizes, int n_in,
                              void* d_out, int out_size, void* d_ws, size_t ws_size,
                              hipStream_t stream) {
    const float* x        = (const float*)d_in[0];
    const int*   eidx     = (const int*)d_in[1];
    const float* eattr    = (const float*)d_in[2];
    /* d_in[3] = batch (unused; graphs are contiguous uniform blocks) */
    const float* Wlin     = (const float*)d_in[4];
    const float* blin     = (const float*)d_in[5];
    const float* Wsrc     = (const float*)d_in[6];
    const float* att_src  = (const float*)d_in[7];
    const float* att_dst  = (const float*)d_in[8];
    const float* Wedge    = (const float*)d_in[9];
    const float* att_edge = (const float*)d_in[10];
    const float* bgat     = (const float*)d_in[11];
    const float* Wgcn     = (const float*)d_in[12];
    const float* bgcn     = (const float*)d_in[13];
    const float* W1       = (const float*)d_in[14];
    const float* b1       = (const float*)d_in[15];
    const float* W2       = (const float*)d_in[16];
    const float* b2       = (const float*)d_in[17];
    const float* W3       = (const float*)d_in[18];
    const float* b3       = (const float*)d_in[19];
    float* out = (float*)d_out;

    const int* srcp = eidx;
    const int* dstp = eidx + EE;

    // ---- bump-allocate workspace (~115 MB) ----
    char* w = (char*)d_ws;
    auto alloc = [&](size_t bytes) -> void* {
        void* p = (void*)w;
        w += (bytes + 255) & ~(size_t)255;
        return p;
    };
    _Float16* h16   = (_Float16*)alloc((size_t)NN * NH2 * 2);
    float* xp       = (float*)alloc((size_t)NN * HCC * 4);
    float* outacc   = (float*)alloc((size_t)NN * HCC * 4);   // becomes x1 in-place
    float* asv      = (float*)alloc((size_t)NN * NHEAD * 4);
    float* adv      = (float*)alloc((size_t)NN * NHEAD * 4);
    float* cnt      = (float*)alloc((size_t)NN * 4);
    float* la       = (float*)alloc((size_t)NN * 2 * 4);
    float* uf       = (float*)alloc(256);
    float* alpha    = (float*)alloc((size_t)(EE + NN) * NHEAD * 4);
    float* mseg     = (float*)alloc((size_t)NN * NHEAD * 4);
    float* denom    = (float*)alloc((size_t)NN * NHEAD * 4);
    float* xw       = (float*)alloc((size_t)NN * 4);
    float* dinv     = (float*)alloc((size_t)NN * 4);
    float* aggv     = (float*)alloc((size_t)NN * 4);
    float* score    = (float*)alloc((size_t)NN * 4);
    float* rbuf     = (float*)alloc((size_t)NB * 2 * HCC * 4);
    (void)ws_size; (void)n_in; (void)in_sizes; (void)out_size;

    const int T = 256;
    auto blocks = [](long long n, int t) { return (unsigned)((n + t - 1) / t); };

    // ---- init accumulators (workspace is poisoned by harness) ----
    k_fill<<<blocks(NN, T), T, 0, stream>>>(cnt, 0.f, NN);
    k_fill<<<blocks(NN * 2, T), T, 0, stream>>>(la, 0.f, NN * 2);
    k_fill<<<blocks(NN * NHEAD, T), T, 0, stream>>>(mseg, -3.0e38f, NN * NHEAD);
    k_fill<<<blocks(NN * NHEAD, T), T, 0, stream>>>(denom, 0.f, NN * NHEAD);
    k_fill<<<blocks((long long)NN * HCC, T), T, 0, stream>>>(outacc, 0.f, NN * HCC);
    k_fill<<<blocks(NN, T), T, 0, stream>>>(aggv, 0.f, NN);

    // ---- 1) lin + ELU (WMMA) : N/16 waves, 8 waves/block ----
    k_lin_elu<<<NN / 16 / 8, 256, 0, stream>>>(x, Wlin, blin, h16);

    // ---- 2) xp = h @ W_src^T (WMMA) : (N/16)*6 waves ----
    k_xp<<<NN / 16 * 6 / 8, 256, 0, stream>>>(h16, Wsrc, xp);

    // ---- 3) node attention scalars ----
    k_attn_node<<<blocks((long long)NN * NHEAD, T), T, 0, stream>>>(xp, att_src, att_dst, asv, adv);

    // ---- 4) self-loop fill ('mean') ----
    k_cnt_loop<<<blocks(EE, T), T, 0, stream>>>(dstp, eattr, cnt, la);
    k_loop_norm<<<blocks(NN, T), T, 0, stream>>>(cnt, la);
    k_uedge<<<1, 32, 0, stream>>>(Wedge, att_edge, uf);

    // ---- 5/6/7) scatter softmax + weighted aggregation ----
    k_alpha<<<blocks(EE + NN, T), T, 0, stream>>>(srcp, dstp, eattr, la, asv, adv, uf, alpha, mseg);
    k_expsum<<<blocks((long long)(EE + NN) * NHEAD, T), T, 0, stream>>>(dstp, mseg, alpha, denom);
    k_agg<<<blocks((long long)(EE + NN) * 48, T), T, 0, stream>>>(srcp, dstp, alpha, denom, xp, outacc);

    // ---- 8) x1 = relu(out + b_gat) ----
    k_x1<<<blocks((long long)NN * HCC, T), T, 0, stream>>>(outacc, bgat);

    // ---- 9) SAGPool GCN score ----
    k_score_prep<<<blocks(NN, T), T, 0, stream>>>(outacc, Wgcn, cnt, xw, dinv);
    k_gcn_scatter<<<blocks(EE, T), T, 0, stream>>>(srcp, dstp, xw, dinv, aggv);
    k_score_final<<<blocks(NN, T), T, 0, stream>>>(aggv, xw, dinv, bgcn, score);

    // ---- 10) top-k mask + gmp/gap readout ----
    k_pool<<<NB, 256, 0, stream>>>(score, outacc, rbuf);

    // ---- 11) MLP + log_softmax ----
    k_mlp<<<NB, 128, 0, stream>>>(rbuf, W1, b1, W2, b2, W3, b3, out);
}
